// Lip2SPRealTime_3745211482820
// MI455X (gfx1250) — matrix operand order repaired and checked by reading, not compiled, not run on hardware
//
#include <hip/hip_runtime.h>
#include <math.h>

// ---------------------------------------------------------------------------
// 2-layer GRU (B=64, H=1024, T=500) + zoneout + output projection (160).
// Strategy: bf16 WMMA (v_wmma_f32_16x16x32_bf16, f32 accumulate), transposed
// bf16-packed weights resident in L2 (192 MB), and ONE persistent kernel for
// the whole recurrence (32 blocks, software grid barrier) instead of 1000
// launches. Per step, each 16x16 gate tile is computed by two waves that
// split K and reduce through LDS, halving the sequential WMMA chain.
// ---------------------------------------------------------------------------
#define BATCH 64
#define HID   1024
#define H3    3072
#define TSTEPS 500
#define OUT2  160
#define BH    (BATCH * HID)   // 65536
#define GRID_BLOCKS 32        // persistent blocks: 32 x 8 waves = 256 waves

typedef __attribute__((ext_vector_type(16))) __bf16 v16bf;
typedef __attribute__((ext_vector_type(8)))  float  v8f;

union TileU { uint4 u[2]; v16bf v; };

__device__ __forceinline__ unsigned short f2bf(float f) {
    unsigned int u = __float_as_uint(f);
    unsigned int r = u + 0x7FFFu + ((u >> 16) & 1u);  // round-to-nearest-even
    return (unsigned short)(r >> 16);
}

// ---- A-tile loader: 16x32 bf16 (MxK), row-major source, ld elements/row ----
// ISA layout: lanes 0-15 -> row M=lane, elems 0..7 = K k0..k0+7, 8..15 = k0+16..23
//             lanes 16-31 -> row M=lane-16, elems = k0+8..15 and k0+24..31
__device__ __forceinline__ v16bf load_a_tile(const unsigned short* X, int ld,
                                             int m0, int k0) {
    int lane = threadIdx.x & 31;
    int m = m0 + (lane & 15);
    int kb = k0 + ((lane >> 4) << 3);
    const unsigned short* p = X + (size_t)m * ld + kb;
    TileU r;
    r.u[0] = *(const uint4*)(p);        // K kb..kb+7
    r.u[1] = *(const uint4*)(p + 16);   // K kb+16..kb+23
    return r.v;
}

// ---- B-tile loader: 32x16 bf16 (KxN) from transposed weights WT[k][n] ------
// ISA layout: lane L holds N col n0+(L&15); elems 0..7 = K kb..kb+7, 8..15 = kb+16..23, kb = k0+8*(L>>4)
__device__ __forceinline__ v16bf load_b_tile(const unsigned short* WT, int ld,
                                             int k0, int n0) {
    int lane = threadIdx.x & 31;
    int kb = k0 + ((lane >> 4) << 3);
    const unsigned short* p = WT + (size_t)kb * ld + n0 + (lane & 15);
    TileU r;
    #pragma unroll
    for (int s = 0; s < 16; ++s)
        r.v[s] = __builtin_bit_cast(__bf16, p[(size_t)(s < 8 ? s : s + 8) * ld]);
    return r.v;
}

__device__ __forceinline__ v8f wmma_bf16(v16bf a, v16bf b, v8f c) {
    return __builtin_amdgcn_wmma_f32_16x16x32_bf16(false, a, false, b,
                                                   (short)0, c, false, false);
}

// ---------------------------------------------------------------------------
// Software grid barrier for the persistent kernel (monotonic counter).
// Release-fence writes, one atomic add per block, acquire-spin by thread 0,
// then an agent-scope acquire fence by every wave so each CU's L0 is
// invalidated before re-reading remote hidden-state writes.
// ---------------------------------------------------------------------------
__device__ __forceinline__ void grid_sync(unsigned int* bar, unsigned int& gen) {
    __builtin_amdgcn_fence(__ATOMIC_RELEASE, "agent");
    __syncthreads();
    if (threadIdx.x == 0) {
        __atomic_fetch_add(bar, 1u, __ATOMIC_RELAXED);
        while (__atomic_load_n(bar, __ATOMIC_ACQUIRE) < gen + GRID_BLOCKS) {
            __builtin_amdgcn_s_sleep(2);
        }
    }
    gen += GRID_BLOCKS;
    __syncthreads();
    __builtin_amdgcn_fence(__ATOMIC_ACQUIRE, "agent");
}

// ---------------------------------------------------------------------------
// One GRU layer-step for this wave's tile pair (2 M-tiles x 1 N-tile), with a
// 2-way K split: kslice 0 -> K[0,512), kslice 1 -> K[512,1024). Slice 1 dumps
// its partial accumulators to LDS; slice 0 reduces and applies the fused
// r/z/n gate math + zoneout in the WMMA C/D register layout.
// ---------------------------------------------------------------------------
__device__ __forceinline__ void layer_step(
    const unsigned short* __restrict__ xb,    // (64,1024) bf16 layer input
    const unsigned short* __restrict__ hbp,   // (64,1024) bf16 h_prev
    const float*          __restrict__ hfp,   // (64,1024) f32  h_prev
    const unsigned short* __restrict__ WTih,  // (1024,3072) bf16
    const unsigned short* __restrict__ WThh,  // (1024,3072) bf16
    const float*          __restrict__ bih,   // (3072)
    const float*          __restrict__ bhh,   // (3072)
    float*                __restrict__ hfn,   // (64,1024) f32  h_new
    unsigned short*       __restrict__ hbn,   // (64,1024) bf16 h_new
    unsigned short*       __restrict__ hbx,   // optional extra bf16 copy (h1 log)
    int m0, int n0, int kslice,
    float* __restrict__ sacc)                 // this pair's 2048-float LDS slab
{
    const int lane = threadIdx.x & 31;

    v8f acc[4][2];   // [r,z,i_n,h_n][mi]
    #pragma unroll
    for (int g = 0; g < 4; ++g)
        #pragma unroll
        for (int mi = 0; mi < 2; ++mi)
            acc[g][mi] = (v8f){};

    const int kbeg = kslice * (HID / 2);
    const int kend = kbeg + (HID / 2);
    #pragma unroll 4
    for (int k0 = kbeg; k0 < kend; k0 += 32) {
        v16bf b_ri = load_b_tile(WTih, H3, k0, n0);
        v16bf b_zi = load_b_tile(WTih, H3, k0, n0 + HID);
        v16bf b_ni = load_b_tile(WTih, H3, k0, n0 + 2 * HID);
        v16bf b_rh = load_b_tile(WThh, H3, k0, n0);
        v16bf b_zh = load_b_tile(WThh, H3, k0, n0 + HID);
        v16bf b_nh = load_b_tile(WThh, H3, k0, n0 + 2 * HID);
        #pragma unroll
        for (int mi = 0; mi < 2; ++mi) {
            v16bf ax = load_a_tile(xb,  HID, m0 + mi * 16, k0);
            v16bf ah = load_a_tile(hbp, HID, m0 + mi * 16, k0);
            acc[0][mi] = wmma_bf16(ax, b_ri, acc[0][mi]);
            acc[0][mi] = wmma_bf16(ah, b_rh, acc[0][mi]);
            acc[1][mi] = wmma_bf16(ax, b_zi, acc[1][mi]);
            acc[1][mi] = wmma_bf16(ah, b_zh, acc[1][mi]);
            acc[2][mi] = wmma_bf16(ax, b_ni, acc[2][mi]);
            acc[3][mi] = wmma_bf16(ah, b_nh, acc[3][mi]);
        }
    }

    // Cross-wave K reduction through LDS (upper slice -> lower slice)
    if (kslice == 1) {
        #pragma unroll
        for (int g = 0; g < 4; ++g)
            #pragma unroll
            for (int mi = 0; mi < 2; ++mi)
                #pragma unroll
                for (int v = 0; v < 8; ++v)
                    sacc[(g * 2 + mi) * 256 + lane * 8 + v] = acc[g][mi][v];
    }
    __syncthreads();

    if (kslice == 0) {
        #pragma unroll
        for (int g = 0; g < 4; ++g)
            #pragma unroll
            for (int mi = 0; mi < 2; ++mi)
                #pragma unroll
                for (int v = 0; v < 8; ++v)
                    acc[g][mi][v] += sacc[(g * 2 + mi) * 256 + lane * 8 + v];

        // Gate math in the C/D register layout:
        //   lanes 0-15:  elem v -> (row m0+v,   col n0+lane)
        //   lanes 16-31: elem v -> (row m0+8+v, col n0+lane-16)
        const int j = n0 + (lane & 15);
        const float br_b = bih[j]           + bhh[j];
        const float bz_b = bih[j + HID]     + bhh[j + HID];
        const float bi_n = bih[j + 2 * HID];
        const float bh_n = bhh[j + 2 * HID];

        #pragma unroll
        for (int mi = 0; mi < 2; ++mi) {
            const int mbase = m0 + mi * 16 + ((lane >> 4) << 3);
            #pragma unroll
            for (int v = 0; v < 8; ++v) {
                const size_t ix = (size_t)(mbase + v) * HID + j;
                float r  = 1.0f / (1.0f + expf(-(acc[0][mi][v] + br_b)));
                float z  = 1.0f / (1.0f + expf(-(acc[1][mi][v] + bz_b)));
                float n  = tanhf(acc[2][mi][v] + bi_n + r * (acc[3][mi][v] + bh_n));
                float hp = hfp[ix];
                float hc = (1.0f - z) * n + z * hp;
                float h  = 0.1f * hp + 0.9f * hc;    // eval-mode zoneout blend
                hfn[ix] = h;
                unsigned short hb16 = f2bf(h);
                hbn[ix] = hb16;
                if (hbx) hbx[ix] = hb16;
            }
        }
    }
}

// ---------------------------------------------------------------------------
// Persistent recurrence kernel: 32 blocks x 256 threads = 256 waves.
// wave -> (tile pair, K slice); 128 tile pairs = 64 N-tiles x 2 M-pairs.
// Two grid barriers per timestep (after layer 0 and after layer 1).
// ---------------------------------------------------------------------------
__global__ __launch_bounds__(256) void gru_persistent(
    const unsigned short* __restrict__ Xt,     // (T,64,1024) bf16
    const unsigned short* __restrict__ WihT,   // (2,1024,3072) bf16
    const unsigned short* __restrict__ WhhT,   // (2,1024,3072) bf16
    const float*          __restrict__ bih,    // (2,3072)
    const float*          __restrict__ bhh,    // (2,3072)
    float*                __restrict__ hf,     // (4,64,1024) f32  [l][parity]
    unsigned short*       __restrict__ hbst,   // (4,64,1024) bf16 [l][parity]
    unsigned short*       __restrict__ Hb1all, // (T,64,1024) bf16 h1 log
    unsigned int*         __restrict__ bar)    // grid-barrier counter
{
    __shared__ float sacc[4 * 2048];   // 32 KB: 4 tile pairs x 2048 f32

    const int wave   = threadIdx.x >> 5;
    const int kslice = wave & 1;
    const int pair   = blockIdx.x * 4 + (wave >> 1);  // 0..127
    const int m0     = (pair & 1) * 32;
    const int n0     = (pair >> 1) * 16;
    float* my_sacc   = sacc + (wave >> 1) * 2048;
    unsigned int gen = 0;

    for (int t = 0; t < TSTEPS; ++t) {
        const int p = t & 1, q = p ^ 1;
        // layer 0: input = Xt[t]
        layer_step(Xt + (size_t)t * BH,
                   hbst + (size_t)p * BH, hf + (size_t)p * BH,
                   WihT, WhhT, bih, bhh,
                   hf + (size_t)q * BH, hbst + (size_t)q * BH,
                   (unsigned short*)nullptr,
                   m0, n0, kslice, my_sacc);
        grid_sync(bar, gen);
        // layer 1: input = layer-0 h_new (bf16); log h1 for output projection
        layer_step(hbst + (size_t)q * BH,
                   hbst + (size_t)(2 + p) * BH, hf + (size_t)(2 + p) * BH,
                   WihT + (size_t)HID * H3, WhhT + (size_t)HID * H3,
                   bih + H3, bhh + H3,
                   hf + (size_t)(2 + q) * BH, hbst + (size_t)(2 + q) * BH,
                   Hb1all + (size_t)t * BH,
                   m0, n0, kslice, my_sacc);
        grid_sync(bar, gen);
    }
}

// ---------------------------------------------------------------------------
// Packing kernels (one-time, fully parallel)
// ---------------------------------------------------------------------------

// (L,3H,H) f32 -> (L,H,3H) bf16 transposed
__global__ void pack_w3(const float* __restrict__ W, unsigned short* __restrict__ WT) {
    int i = blockIdx.x * blockDim.x + threadIdx.x;
    const int per = H3 * HID;
    if (i >= 2 * per) return;
    int l = i / per, rem = i % per;
    int j = rem / HID, k = rem % HID;
    WT[(size_t)l * per + (size_t)k * H3 + j] = f2bf(W[i]);
}

// (160,1024) f32 -> (1024,160) bf16
__global__ void pack_wout(const float* __restrict__ W, unsigned short* __restrict__ WT) {
    int i = blockIdx.x * blockDim.x + threadIdx.x;
    if (i >= OUT2 * HID) return;
    int j = i / HID, k = i % HID;
    WT[(size_t)k * OUT2 + j] = f2bf(W[i]);
}

// res_output (B,H,T) f32 -> Xt (T,B,H) bf16
__global__ void pack_x(const float* __restrict__ X, unsigned short* __restrict__ Xt) {
    size_t i = (size_t)blockIdx.x * blockDim.x + threadIdx.x;
    if (i >= (size_t)TSTEPS * BH) return;
    int t = (int)(i / BH);
    int rem = (int)(i % BH);
    int b = rem / HID, k = rem % HID;
    Xt[i] = f2bf(X[(size_t)b * HID * TSTEPS + (size_t)k * TSTEPS + t]);
}

// zero hidden-state buffers + grid-barrier counter (runs every launch ->
// deterministic across graph replays)
__global__ void init_state(float* __restrict__ hf, unsigned short* __restrict__ hb,
                           unsigned int* __restrict__ bar) {
    int i = blockIdx.x * blockDim.x + threadIdx.x;
    if (i < 4 * BH) { hf[i] = 0.0f; hb[i] = 0; }
    if (i == 0) *bar = 0u;
}

// ---------------------------------------------------------------------------
// Output projection: Y = H1_all(32000x1024) @ WoutT(1024x160) + bout,
// scattered into out (64, 80, 1000) per the reduction-factor reshape.
// 2000 M-tiles x 10 N-tiles; 1 tile per wave; grid 2500 x 256.
// ---------------------------------------------------------------------------
__global__ __launch_bounds__(256) void out_proj(
    const unsigned short* __restrict__ Hb,     // (T*B, 1024) bf16
    const unsigned short* __restrict__ WoutT,  // (1024, 160) bf16
    const float*          __restrict__ bout,   // (160)
    float*                __restrict__ out)    // (64, 80, 1000)
{
    const int wid   = blockIdx.x * 8 + (threadIdx.x >> 5);
    const int ntile = wid % 10;
    const int mtile = wid / 10;
    const int m0 = mtile * 16, n0 = ntile * 16;
    const int lane = threadIdx.x & 31;

    v8f acc = {};
    #pragma unroll 4
    for (int k0 = 0; k0 < HID; k0 += 32) {
        v16bf a = load_a_tile(Hb, HID, m0, k0);
        v16bf b = load_b_tile(WoutT, OUT2, k0, n0);
        acc = wmma_bf16(a, b, acc);
    }

    const int j  = n0 + (lane & 15);
    const float bj = bout[j];
    const int mbase = m0 + ((lane >> 4) << 3);
    #pragma unroll
    for (int v = 0; v < 8; ++v) {
        const int m = mbase + v;          // m = t*64 + b
        const int t = m >> 6, b = m & 63;
        // ys.reshape(T,B,OC,2) -> out[b][oc][t*2+rf], oc=j>>1, rf=j&1
        out[(size_t)b * 80 * 1000 + (size_t)(j >> 1) * 1000 + t * 2 + (j & 1)]
            = acc[v] + bj;
    }
}

// ---------------------------------------------------------------------------
extern "C" void kernel_launch(void* const* d_in, const int* in_sizes, int n_in,
                              void* d_out, int out_size, void* d_ws, size_t ws_size,
                              hipStream_t stream) {
    const float* res  = (const float*)d_in[0];  // (64,1024,500)
    const float* Wih  = (const float*)d_in[1];  // (2,3072,1024)
    const float* Whh  = (const float*)d_in[2];  // (2,3072,1024)
    const float* bih  = (const float*)d_in[3];  // (2,3072)
    const float* bhh  = (const float*)d_in[4];  // (2,3072)
    const float* Wout = (const float*)d_in[5];  // (160,1024)
    const float* bout = (const float*)d_in[6];  // (160)
    float* out = (float*)d_out;                 // (64,80,1000)

    // Workspace carve-up (~159 MB total)
    char* ws = (char*)d_ws;
    size_t off = 0;
    auto carve = [&](size_t bytes) -> void* {
        void* p = ws + off;
        off = (off + bytes + 255) & ~(size_t)255;
        return p;
    };
    unsigned short* WihT   = (unsigned short*)carve((size_t)2 * HID * H3 * 2);
    unsigned short* WhhT   = (unsigned short*)carve((size_t)2 * HID * H3 * 2);
    unsigned short* WoutT  = (unsigned short*)carve((size_t)HID * OUT2 * 2);
    unsigned short* Xt     = (unsigned short*)carve((size_t)TSTEPS * BH * 2);
    unsigned short* Hb1all = (unsigned short*)carve((size_t)TSTEPS * BH * 2);
    float*          hf     = (float*)carve((size_t)4 * BH * 4);   // [l][parity]
    unsigned short* hbst   = (unsigned short*)carve((size_t)4 * BH * 2);
    unsigned int*   bar    = (unsigned int*)carve(256);

    // One-time packing / init
    {
        int n = 2 * H3 * HID;
        pack_w3<<<(n + 255) / 256, 256, 0, stream>>>(Wih, WihT);
        pack_w3<<<(n + 255) / 256, 256, 0, stream>>>(Whh, WhhT);
    }
    pack_wout<<<(OUT2 * HID + 255) / 256, 256, 0, stream>>>(Wout, WoutT);
    {
        size_t n = (size_t)TSTEPS * BH;
        pack_x<<<(unsigned)((n + 255) / 256), 256, 0, stream>>>(res, Xt);
    }
    init_state<<<(4 * BH + 255) / 256, 256, 0, stream>>>(hf, hbst, bar);

    // Entire 500-step, 2-layer recurrence in ONE persistent kernel
    gru_persistent<<<GRID_BLOCKS, 256, 0, stream>>>(
        Xt, WihT, WhhT, bih, bhh, hf, hbst, Hb1all, bar);

    // Output projection + reduction-factor scatter
    out_proj<<<2500, 256, 0, stream>>>(Hb1all, WoutT, bout, out);
}
